// GPT_21019569946962
// MI455X (gfx1250) — matrix operand order repaired and checked by reading, not compiled, not run on hardware
//
#include <hip/hip_runtime.h>
#include <hip/hip_bf16.h>
#include <math.h>

// ---------------------------------------------------------------------------
// GPT-2 small forward pass for gfx1250 (MI455X), bf16 WMMA with f32 accumulate.
// Round 2: + Tensor Data Mover (tensor_load_to_lds) staging with LDS padding,
//          + double-buffered GEMM K-loop (TDM overlaps WMMA compute),
//          + packed B-tile staging, vectorized V-transpose staging.
// ---------------------------------------------------------------------------

typedef unsigned short u16;
typedef __attribute__((ext_vector_type(16))) __bf16 v16bf;
typedef __attribute__((ext_vector_type(8)))  float  v8f;
typedef __attribute__((ext_vector_type(4)))  unsigned int v4u;
typedef __attribute__((ext_vector_type(8)))  int v8i;
typedef __attribute__((ext_vector_type(4)))  int v4i;

union Frag {
    v16bf bf;
    uint4 q[2];
};

#define DMODEL 768
#define DFF    3072
#define LSEQ   1024
#define NHEADS 12
#define ROWS   2048      /* B*L */
#define HDIM   64

// --- TDM availability (device pass only; host parse uses fallback path) ----
#if defined(__has_builtin)
#  if __has_builtin(__builtin_amdgcn_tensor_load_to_lds) && \
      __has_builtin(__builtin_amdgcn_s_wait_tensorcnt)
#    define HAVE_TDM 1
#  endif
#endif
#ifndef HAVE_TDM
#  define HAVE_TDM 0
#endif

// fp32 -> bf16 with round-to-nearest-even (bit pattern as u16)
__device__ inline u16 f2bf(float f) {
    union { float f; unsigned u; } c; c.f = f;
    unsigned r = c.u + 0x7FFFu + ((c.u >> 16) & 1u);
    return (u16)(r >> 16);
}

__device__ inline v8f v8zero() { v8f z = {0.f,0.f,0.f,0.f,0.f,0.f,0.f,0.f}; return z; }

__device__ inline v8f wmma_bf16(Frag a, Frag b, v8f c) {
    // D = A(16x32 bf16) * B(32x16 bf16) + C(16x16 f32)
    return __builtin_amdgcn_wmma_f32_16x16x32_bf16(
        /*neg_a=*/false, a.bf, /*neg_b=*/false, b.bf,
        /*c_mod=*/(short)0, c, /*reuse_a=*/false, /*reuse_b=*/false);
}

#if HAVE_TDM
// ---------------------------------------------------------------------------
// Tensor DMA: load a 2D bf16 tile (tile0 x tile1 elements, row stride stride0
// elements) from global into LDS at ldsOff, inserting padAmt+1 DWORDs of LDS
// padding after every 2^(padIvl+1) DWORDs (CDNA5 ISA D# layout).
// ---------------------------------------------------------------------------
__device__ inline void tdm_load2d_bf16(const u16* g, unsigned ldsOff,
                                       unsigned tile0, unsigned tile1,
                                       unsigned stride0,
                                       unsigned tdim0, unsigned tdim1,
                                       unsigned padIvl, unsigned padAmt) {
    unsigned long long ga = (unsigned long long)(uintptr_t)g;
    // D# group 0: count=1 | lds_addr | global_addr[56:0] | type=2
    v4u g0 = { 1u,
               ldsOff,
               (unsigned)(ga & 0xFFFFFFFFu),
               (unsigned)((ga >> 32) & 0x01FFFFFFu) | (2u << 30) };
    // D# group 1: data_size=2B, pad cfg, tensor/tile dims, dim0 stride
    unsigned w0 = (1u << 16) | (1u << 20) | (padIvl << 22) | (padAmt << 25);
    v8i g1 = { (int)w0,
               (int)((tdim0 & 0xFFFFu) << 16),
               (int)(((tdim0 >> 16) & 0xFFFFu) | ((tdim1 & 0xFFFFu) << 16)),
               (int)(((tdim1 >> 16) & 0xFFFFu) | ((tile0 & 0xFFFFu) << 16)),
               (int)(tile1 & 0xFFFFu),
               (int)stride0,
               0, 0 };
    v4i gz = { 0, 0, 0, 0 };
#if defined(__clang_major__) && __clang_major__ >= 23
    v8i gz8 = { 0, 0, 0, 0, 0, 0, 0, 0 };
    __builtin_amdgcn_tensor_load_to_lds(g0, g1, gz, gz, gz8, 0);
#else
    __builtin_amdgcn_tensor_load_to_lds(g0, g1, gz, gz, 0);
#endif
}
#endif // HAVE_TDM

// ---------------------------------------------------------------------------
// Embedding gather: x[row, :] = tok_emb[ids[row], :] + pos_emb[row % L, :]
// ---------------------------------------------------------------------------
__global__ __launch_bounds__(256)
void embed_kernel(const int* __restrict__ ids, const float* __restrict__ tok,
                  const float* __restrict__ pos, float* __restrict__ x) {
    int row = blockIdx.x;
    int l   = row & (LSEQ - 1);
    int t   = ids[row];
    for (int c = threadIdx.x; c < DMODEL; c += 256)
        x[(size_t)row * DMODEL + c] =
            tok[(size_t)t * DMODEL + c] + pos[(size_t)l * DMODEL + c];
}

// ---------------------------------------------------------------------------
// LayerNorm (one block per row of 768): fp32 in, bf16 out
// ---------------------------------------------------------------------------
__global__ __launch_bounds__(256)
void layernorm_kernel(const float* __restrict__ x, const float* __restrict__ sc,
                      const float* __restrict__ bi, u16* __restrict__ out) {
    int row = blockIdx.x, tid = threadIdx.x;
    const float* xr = x + (size_t)row * DMODEL;
    float v0 = xr[tid], v1 = xr[tid + 256], v2 = xr[tid + 512];

    __shared__ float red[8];
    float s = v0 + v1 + v2;
    #pragma unroll
    for (int m = 16; m > 0; m >>= 1) s += __shfl_xor(s, m, 32);
    if ((tid & 31) == 0) red[tid >> 5] = s;
    __syncthreads();
    float tot = 0.f;
    #pragma unroll
    for (int i = 0; i < 8; ++i) tot += red[i];
    float mu = tot * (1.0f / DMODEL);

    float d0 = v0 - mu, d1 = v1 - mu, d2 = v2 - mu;
    float q = d0 * d0 + d1 * d1 + d2 * d2;
    __syncthreads();
    #pragma unroll
    for (int m = 16; m > 0; m >>= 1) q += __shfl_xor(q, m, 32);
    if ((tid & 31) == 0) red[tid >> 5] = q;
    __syncthreads();
    float tv = 0.f;
    #pragma unroll
    for (int i = 0; i < 8; ++i) tv += red[i];
    float inv = rsqrtf(tv * (1.0f / DMODEL) + 1e-6f);

    out[(size_t)row * DMODEL + tid      ] = f2bf(d0 * inv * sc[tid      ] + bi[tid      ]);
    out[(size_t)row * DMODEL + tid + 256] = f2bf(d1 * inv * sc[tid + 256] + bi[tid + 256]);
    out[(size_t)row * DMODEL + tid + 512] = f2bf(d2 * inv * sc[tid + 512] + bi[tid + 512]);
}

// ---------------------------------------------------------------------------
// GEMM: C[M,N] = A[M,K](bf16) * W[K,N](fp32->bf16 during staging) (+bias)
// (+tanh-GELU) (+residual). Block tile 128x128, BK=32, 8 waves as 2(M)x4(N),
// each wave does a 64x32 tile = 4x2 WMMA 16x16x32 bf16 accumulators.
// Double-buffered: A tile via TDM (async DMA, LDS pad -> stride GSTR),
// B tile via fp32->bf16 convert+transpose stores.
// ---------------------------------------------------------------------------
#define GSTR 40   /* LDS row stride in u16: 16 DW data + 4 DW pad */
#define ASZ  (128 * GSTR)

template <bool BIAS, bool GELU, bool RES, bool BF16OUT>
__global__ __launch_bounds__(256)
void gemm_wmma_kernel(const u16* __restrict__ A, const float* __restrict__ Bw,
                      const float* __restrict__ bias, const float* __restrict__ Res,
                      float* __restrict__ Cf, u16* __restrict__ Cb,
                      int N, int K) {
    __shared__ u16 As[2][ASZ];
    __shared__ u16 Bs[2][ASZ];

    const int tid  = threadIdx.x;
    const int w    = tid >> 5, lane = tid & 31;
    const int lm   = lane & 15, hv = lane >> 4;
    const int wm   = w >> 2,  wn = w & 3;     // 2x4 wave grid
    const int m0   = blockIdx.y * 128, n0 = blockIdx.x * 128;

    v8f acc[4][2];
    #pragma unroll
    for (int mt = 0; mt < 4; ++mt)
        #pragma unroll
        for (int nt = 0; nt < 2; ++nt) acc[mt][nt] = v8zero();

    const int arow = tid >> 1, acol = (tid & 1) * 16;   // manual A stage mapping
    const int bkp  = tid >> 4, bnp  = (tid & 15) * 8;   // B stage: k-pair x 8 n

    // --- stage A tile kt into buffer b ---
    auto stageA = [&](int kt, int b) {
#if HAVE_TDM
        if (tid == 0)
            tdm_load2d_bf16(A + (size_t)m0 * K + kt,
                            (unsigned)(uintptr_t)&As[b][0],
                            /*tile0=*/32, /*tile1=*/128, /*stride0=*/(unsigned)K,
                            /*tdim0=*/(unsigned)K, /*tdim1=*/0x40000u,
                            /*padIvl(16DW)=*/3, /*padAmt(4DW)=*/3);
#else
        const u16* src = A + (size_t)(m0 + arow) * K + kt + acol;
        uint4 x0 = ((const uint4*)src)[0];
        uint4 x1 = ((const uint4*)src)[1];
        *(uint4*)&As[b][arow * GSTR + acol]     = x0;
        *(uint4*)&As[b][arow * GSTR + acol + 8] = x1;
#endif
    };
    // --- stage B tile kt into buffer b: Bs[n][k], packed pair of k per b32 ---
    auto stageB = [&](int kt, int b) {
        const float* s0 = Bw + (size_t)(kt + 2 * bkp) * N + n0 + bnp;
        const float* s1 = s0 + N;
        if (kt + 32 < K)
            __builtin_prefetch(s0 + (size_t)32 * N, 0, 1);
        #pragma unroll
        for (int i = 0; i < 8; ++i) {
            unsigned lo = f2bf(s0[i]);
            unsigned hi = f2bf(s1[i]);
            *(unsigned*)&Bs[b][(bnp + i) * GSTR + 2 * bkp] = lo | (hi << 16);
        }
    };

    stageA(0, 0);
    stageB(0, 0);

    for (int kt = 0; kt < K; kt += 32) {
        const int buf = (kt >> 5) & 1;
#if HAVE_TDM
        if (tid == 0) __builtin_amdgcn_s_wait_tensorcnt(0);
#endif
        __syncthreads();                       // tile kt resident & visible
        if (kt + 32 < K) {                     // overlap next-tile staging
            stageA(kt + 32, buf ^ 1);
            stageB(kt + 32, buf ^ 1);
        }

        Frag af[4], bfg[2];
        #pragma unroll
        for (int mt = 0; mt < 4; ++mt) {
            const u16* p = &As[buf][(wm * 64 + mt * 16 + lm) * GSTR + hv * 8];
            af[mt].q[0] = *(const uint4*)p;
            af[mt].q[1] = *(const uint4*)(p + 16);
        }
        #pragma unroll
        for (int nt = 0; nt < 2; ++nt) {
            const u16* p = &Bs[buf][(wn * 32 + nt * 16 + lm) * GSTR + hv * 8];
            bfg[nt].q[0] = *(const uint4*)p;
            bfg[nt].q[1] = *(const uint4*)(p + 16);
        }
        #pragma unroll
        for (int mt = 0; mt < 4; ++mt)
            #pragma unroll
            for (int nt = 0; nt < 2; ++nt)
                acc[mt][nt] = wmma_bf16(af[mt], bfg[nt], acc[mt][nt]);
    }

    // --- epilogue ---
    #pragma unroll
    for (int mt = 0; mt < 4; ++mt) {
        int row = m0 + wm * 64 + mt * 16 + hv * 8;
        #pragma unroll
        for (int nt = 0; nt < 2; ++nt) {
            int col = n0 + wn * 32 + nt * 16 + lm;
            float bval = BIAS ? bias[col] : 0.f;
            #pragma unroll
            for (int r = 0; r < 8; ++r) {
                float v = acc[mt][nt][r] + bval;
                if (GELU) {
                    float t = tanhf(0.7978845608028654f * (v + 0.044715f * v * v * v));
                    v = 0.5f * v * (1.f + t);
                }
                size_t idx = (size_t)(row + r) * N + col;
                if (RES) v += Res[idx];
                if (BF16OUT) Cb[idx] = f2bf(v);
                else         Cf[idx] = v;
            }
        }
    }
}

// ---------------------------------------------------------------------------
// Causal flash attention (bf16 WMMA, f32 online softmax).
// Block = 128 threads (4 waves); wave w owns 16 query rows; block covers 64.
// K tile staged by TDM (row-major = B-frag layout); V staged transposed;
// P round-trips LDS to convert C-frag -> A-frag.  grid = (L/64, NHEADS, B)
// ---------------------------------------------------------------------------
#define FSTR 72   /* 32 DW data + 4 DW pad per row */

__global__ __launch_bounds__(128)
void flash_attn_kernel(const u16* __restrict__ Qg, const u16* __restrict__ Kg,
                       const u16* __restrict__ Vg, u16* __restrict__ Og) {
    __shared__ u16 Ks[64 * FSTR];          // Ks[key][hd]
    __shared__ u16 Vt[64 * FSTR];          // Vt[hd][key]
    __shared__ u16 Ps[4 * 16 * FSTR];      // per-wave P rows

    const int tid = threadIdx.x;
    const int w = tid >> 5, lane = tid & 31;
    const int lm = lane & 15, hv = lane >> 4;
    const int br = blockIdx.x, h = blockIdx.y, b = blockIdx.z;
    const int r0 = br * 64 + w * 16;                       // wave's first query row
    const size_t base = (size_t)b * LSEQ * DMODEL + h * HDIM;
#if HAVE_TDM
    const unsigned ksOff = (unsigned)(uintptr_t)&Ks[0];
#endif

    // Q fragments (kept in registers for whole kernel)
    Frag qf[2];
    {
        const u16* qrow = Qg + base + (size_t)(r0 + lm) * DMODEL;
        #pragma unroll
        for (int s = 0; s < 2; ++s) {
            qf[s].q[0] = *(const uint4*)(qrow + s * 32 + hv * 8);
            qf[s].q[1] = *(const uint4*)(qrow + s * 32 + 16 + hv * 8);
        }
    }

    v8f oacc[4];
    #pragma unroll
    for (int t = 0; t < 4; ++t) oacc[t] = v8zero();
    float mrow[8], lrow[8];
    #pragma unroll
    for (int r = 0; r < 8; ++r) { mrow[r] = -INFINITY; lrow[r] = 0.f; }

    const int ntiles = br + 1;   // causal: key tiles jt*64 <= last row of block
    for (int jt = 0; jt < ntiles; ++jt) {
        // stage K tile: TDM async DMA (pad 4 DW after each 32 DW row -> FSTR)
#if HAVE_TDM
        if (tid == 0)
            tdm_load2d_bf16(Kg + base + (size_t)(jt * 64) * DMODEL, ksOff,
                            /*tile0=*/64, /*tile1=*/64, /*stride0=*/DMODEL,
                            /*tdim0=*/DMODEL, /*tdim1=*/LSEQ,
                            /*padIvl(32DW)=*/4, /*padAmt(4DW)=*/3);
#else
        for (int idx = tid; idx < 512; idx += 128) {
            int key = idx >> 3, part = idx & 7;
            const uint4* src =
                (const uint4*)(Kg + base + (size_t)(jt * 64 + key) * DMODEL) + part;
            *(uint4*)&Ks[key * FSTR + part * 8] = *src;
        }
#endif
        // stage V tile transposed: vectorized b128 loads, scattered b16 stores
        for (int idx = tid; idx < 512; idx += 128) {
            int key = idx >> 3, hd0 = (idx & 7) * 8;
            uint4 d = *(const uint4*)(Vg + base + (size_t)(jt * 64 + key) * DMODEL + hd0);
            unsigned vals[4] = { d.x, d.y, d.z, d.w };
            #pragma unroll
            for (int i = 0; i < 4; ++i) {
                Vt[(hd0 + 2 * i    ) * FSTR + key] = (u16)(vals[i] & 0xFFFFu);
                Vt[(hd0 + 2 * i + 1) * FSTR + key] = (u16)(vals[i] >> 16);
            }
        }
#if HAVE_TDM
        if (tid == 0) __builtin_amdgcn_s_wait_tensorcnt(0);
#endif
        __syncthreads();

        // S = Q*K^T, scale, causal mask -------------------------------------
        float pvv[4][8], rmax[8];
        #pragma unroll
        for (int r = 0; r < 8; ++r) rmax[r] = -INFINITY;
        #pragma unroll
        for (int nt = 0; nt < 4; ++nt) {
            v8f sa = v8zero();
            #pragma unroll
            for (int s = 0; s < 2; ++s) {
                Frag kf;
                const u16* kp = &Ks[(nt * 16 + lm) * FSTR + s * 32 + hv * 8];
                kf.q[0] = *(const uint4*)kp;
                kf.q[1] = *(const uint4*)(kp + 16);
                sa = wmma_bf16(qf[s], kf, sa);
            }
            int key = jt * 64 + nt * 16 + lm;
            #pragma unroll
            for (int r = 0; r < 8; ++r) {
                int l = r0 + r + hv * 8;
                float v = sa[r] * 0.125f;           // 1/sqrt(64)
                if (key > l) v = -INFINITY;
                pvv[nt][r] = v;
                rmax[r] = fmaxf(rmax[r], v);
            }
        }
        #pragma unroll
        for (int r = 0; r < 8; ++r)
            #pragma unroll
            for (int m = 1; m < 16; m <<= 1)
                rmax[r] = fmaxf(rmax[r], __shfl_xor(rmax[r], m, 32));

        // online softmax update ---------------------------------------------
        float corr[8], rsum[8];
        #pragma unroll
        for (int r = 0; r < 8; ++r) {
            float mn = fmaxf(mrow[r], rmax[r]);
            corr[r] = __expf(mrow[r] - mn);
            mrow[r] = mn;
            rsum[r] = 0.f;
        }
        #pragma unroll
        for (int nt = 0; nt < 4; ++nt)
            #pragma unroll
            for (int r = 0; r < 8; ++r) {
                float p = __expf(pvv[nt][r] - mrow[r]);
                rsum[r] += p;
                Ps[(w * 16 + r + hv * 8) * FSTR + nt * 16 + lm] = f2bf(p);
            }
        #pragma unroll
        for (int r = 0; r < 8; ++r) {
            #pragma unroll
            for (int m = 1; m < 16; m <<= 1) rsum[r] += __shfl_xor(rsum[r], m, 32);
            lrow[r] = lrow[r] * corr[r] + rsum[r];
        }
        #pragma unroll
        for (int t = 0; t < 4; ++t)
            #pragma unroll
            for (int r = 0; r < 8; ++r) oacc[t][r] *= corr[r];
        __syncthreads();   // make P visible / done with Ks reads

        // O += P * V ---------------------------------------------------------
        #pragma unroll
        for (int s = 0; s < 2; ++s) {
            Frag pf;
            const u16* pp = &Ps[(w * 16 + lm) * FSTR + s * 32 + hv * 8];
            pf.q[0] = *(const uint4*)pp;
            pf.q[1] = *(const uint4*)(pp + 16);
            #pragma unroll
            for (int ot = 0; ot < 4; ++ot) {
                Frag vf;
                const u16* vp = &Vt[(ot * 16 + lm) * FSTR + s * 32 + hv * 8];
                vf.q[0] = *(const uint4*)vp;
                vf.q[1] = *(const uint4*)(vp + 16);
                oacc[ot] = wmma_bf16(pf, vf, oacc[ot]);
            }
        }
        __syncthreads();   // before next tile overwrites Ks/Vt
    }

    // normalize and write O (bf16) ------------------------------------------
    #pragma unroll
    for (int ot = 0; ot < 4; ++ot)
        #pragma unroll
        for (int r = 0; r < 8; ++r) {
            int l = r0 + r + hv * 8;
            float o = oacc[ot][r] / lrow[r];
            Og[base + (size_t)l * DMODEL + ot * 16 + lm] = f2bf(o);
        }
}

// ---------------------------------------------------------------------------
// Launch: embed -> 6x(LN, QKV, attn, WO+res, LN, FF1(gelu), FF2+res) -> LN -> head
// ---------------------------------------------------------------------------
extern "C" void kernel_launch(void* const* d_in, const int* in_sizes, int n_in,
                              void* d_out, int out_size, void* d_ws, size_t ws_size,
                              hipStream_t stream) {
    (void)in_sizes; (void)n_in; (void)out_size; (void)ws_size;

    const int*   ids  = (const int*)  d_in[0];
    const float* tok  = (const float*)d_in[1];
    const float* pos  = (const float*)d_in[2];
    const float* ln1s = (const float*)d_in[3];
    const float* ln1b = (const float*)d_in[4];
    const float* wq   = (const float*)d_in[5];
    const float* wk   = (const float*)d_in[6];
    const float* wv   = (const float*)d_in[7];
    const float* wo   = (const float*)d_in[8];
    const float* ln2s = (const float*)d_in[9];
    const float* ln2b = (const float*)d_in[10];
    const float* w1k  = (const float*)d_in[11];
    const float* w1b  = (const float*)d_in[12];
    const float* w2k  = (const float*)d_in[13];
    const float* w2b  = (const float*)d_in[14];
    const float* lnfs = (const float*)d_in[15];
    const float* lnfb = (const float*)d_in[16];
    const float* head = (const float*)d_in[17];
    float* out = (float*)d_out;

    size_t off = 0;
    auto alloc = [&](size_t bytes) {
        void* p = (char*)d_ws + off;
        off += (bytes + 255) & ~(size_t)255;
        return p;
    };
    float* x   = (float*)alloc((size_t)ROWS * DMODEL * 4);
    u16*   lnb = (u16*)  alloc((size_t)ROWS * DMODEL * 2);
    u16*   qb  = (u16*)  alloc((size_t)ROWS * DMODEL * 2);
    u16*   kb  = (u16*)  alloc((size_t)ROWS * DMODEL * 2);
    u16*   vb  = (u16*)  alloc((size_t)ROWS * DMODEL * 2);
    u16*   att = (u16*)  alloc((size_t)ROWS * DMODEL * 2);
    u16*   ffh = (u16*)  alloc((size_t)ROWS * DFF * 2);

    const dim3 blk256(256), blk128(128);
    const dim3 gRows(ROWS);
    const dim3 gD(DMODEL / 128, ROWS / 128);       // (6,16)
    const dim3 gFF(DFF / 128, ROWS / 128);         // (24,16)
    const dim3 gHead(50304 / 128, ROWS / 128);     // (393,16)
    const dim3 gAtt(LSEQ / 64, NHEADS, 2);         // (16,12,2)

    embed_kernel<<<gRows, blk256, 0, stream>>>(ids, tok, pos, x);

    for (int i = 0; i < 6; ++i) {
        layernorm_kernel<<<gRows, blk256, 0, stream>>>(x, ln1s + i * DMODEL,
                                                       ln1b + i * DMODEL, lnb);
        const float* Wq = wq + (size_t)i * DMODEL * DMODEL;
        const float* Wk = wk + (size_t)i * DMODEL * DMODEL;
        const float* Wv = wv + (size_t)i * DMODEL * DMODEL;
        const float* Wo = wo + (size_t)i * DMODEL * DMODEL;
        gemm_wmma_kernel<false,false,false,true><<<gD, blk256, 0, stream>>>(
            lnb, Wq, nullptr, nullptr, nullptr, qb, DMODEL, DMODEL);
        gemm_wmma_kernel<false,false,false,true><<<gD, blk256, 0, stream>>>(
            lnb, Wk, nullptr, nullptr, nullptr, kb, DMODEL, DMODEL);
        gemm_wmma_kernel<false,false,false,true><<<gD, blk256, 0, stream>>>(
            lnb, Wv, nullptr, nullptr, nullptr, vb, DMODEL, DMODEL);

        flash_attn_kernel<<<gAtt, blk128, 0, stream>>>(qb, kb, vb, att);

        // x = x + att @ Wo
        gemm_wmma_kernel<false,false,true,false><<<gD, blk256, 0, stream>>>(
            att, Wo, nullptr, x, x, nullptr, DMODEL, DMODEL);

        layernorm_kernel<<<gRows, blk256, 0, stream>>>(x, ln2s + i * DMODEL,
                                                       ln2b + i * DMODEL, lnb);
        // ffh = gelu(lnb @ W1 + b1)
        gemm_wmma_kernel<true,true,false,true><<<gFF, blk256, 0, stream>>>(
            lnb, w1k + (size_t)i * DMODEL * DFF, w1b + (size_t)i * DFF,
            nullptr, nullptr, ffh, DFF, DMODEL);
        // x = x + ffh @ W2 + b2
        gemm_wmma_kernel<true,false,true,false><<<gD, blk256, 0, stream>>>(
            ffh, w2k + (size_t)i * DFF * DMODEL, w2b + (size_t)i * DMODEL,
            x, x, nullptr, DMODEL, DFF);
    }

    layernorm_kernel<<<gRows, blk256, 0, stream>>>(x, lnfs, lnfb, lnb);
    gemm_wmma_kernel<false,false,false,false><<<gHead, blk256, 0, stream>>>(
        lnb, head, nullptr, nullptr, out, nullptr, 50304, DMODEL);
}
